// _HFMoE_66760971649155
// MI455X (gfx1250) — compile-verified, matrix operand order checked by reading, not run on hardware
//
#include <hip/hip_runtime.h>
#include <hip/hip_bf16.h>

// ---------------------------------------------------------------------------
// MoE (8 experts, top-2, + shared expert) forward for MI455X / gfx1250.
// fp32 weights streamed once from HBM, converted to bf16 on the fly in LDS,
// computed with v_wmma_f32_16x16x32_bf16 (wave32 WMMA). Global->register
// prefetch pipeline hides HBM latency under the WMMA stream.
// ---------------------------------------------------------------------------

typedef __bf16 bf16_t;
typedef bf16_t v16bf __attribute__((ext_vector_type(16)));
typedef float  v8f   __attribute__((ext_vector_type(8)));

#define T_TOK 2048   // B*S
#define HID   2048   // hidden size (GEMM-K for gate/up)
#define NEXP  8
#define IEXP  1408   // per-expert intermediate
#define ISH   5632   // shared-expert intermediate

#define BM  128      // block tile M (tokens)
#define BN  64       // block tile N
#define KC  32       // K step (= WMMA bf16 K)
#define LDT 40       // LDS row stride in bf16 elems (32 data + 8 pad -> 80B rows)

// ---- fp32 -> packed bf16 pair ---------------------------------------------

static __device__ __forceinline__ unsigned int pack_bf16_pair(float a, float b) {
#if __has_builtin(__builtin_amdgcn_cvt_pk_bf16_f32)
  typedef bf16_t v2bf __attribute__((ext_vector_type(2)));
  v2bf p = __builtin_amdgcn_cvt_pk_bf16_f32(a, b);   // v_cvt_pk_bf16_f32
  unsigned int u;
  __builtin_memcpy(&u, &p, 4);
  return u;
#else
  unsigned int ua = __float_as_uint(a);
  unsigned int ub = __float_as_uint(b);
  unsigned int ra = ua + 0x7fffu + ((ua >> 16) & 1u);  // RNE bias
  unsigned int rb = ub + 0x7fffu + ((ub >> 16) & 1u);
  return __builtin_amdgcn_perm(rb, ra, 0x07060302u);   // {rb.hi, ra.hi}
#endif
}

static __device__ __forceinline__ void store_bf4(unsigned short* p, float4 v) {
  uint2 q;
  q.x = pack_bf16_pair(v.x, v.y);
  q.y = pack_bf16_pair(v.z, v.w);
  *(uint2*)p = q;                                      // 8B LDS store
}

union FragU { v16bf v; uint4 q[2]; };

// 16x32 bf16 A/B fragment from a padded LDS tile (row-major, stride LDT).
// Lane l holds row (l&15); lanes 0-15 carry K={0..7,16..23}, lanes 16-31
// carry K={8..15,24..31} per the CDNA5 16-bit matrix layout.
static __device__ __forceinline__ v16bf load_frag(const unsigned short* row0,
                                                  int lane) {
  const unsigned short* p = row0 + (lane & 15) * LDT + ((lane >> 4) << 3);
  FragU f;
  f.q[0] = *(const uint4*)(p);        // K 0..7   (or 8..15)
  f.q[1] = *(const uint4*)(p + 16);   // K 16..23 (or 24..31)
  return f.v;
}

static __device__ __forceinline__ v8f wmma_bf16(v16bf a, v16bf b, v8f c) {
  return __builtin_amdgcn_wmma_f32_16x16x32_bf16(false, a, false, b,
                                                 (short)0, c, false, false);
}

static __device__ __forceinline__ float fast_silu(float g) {
  // g * sigmoid(g) with v_rcp_f32 (plenty accurate next to bf16 matmul error)
  return g * __builtin_amdgcn_rcpf(1.0f + __expf(-g));
}

// ---- router: logits -> softmax -> top-2 -> renorm -> dense combine --------

__global__ __launch_bounds__(256)
void moe_router(const float* __restrict__ x, const float* __restrict__ gw,
                float* __restrict__ combine) {
  const int tok  = (blockIdx.x * blockDim.x + threadIdx.x) >> 5;  // wave/token
  const int lane = threadIdx.x & 31;
  if (tok >= T_TOK) return;

  const float* xr = x + (size_t)tok * HID;
  float acc[NEXP];
#pragma unroll
  for (int e = 0; e < NEXP; ++e) acc[e] = 0.0f;

  for (int h = lane; h < HID; h += 32) {
    float xv = xr[h];
#pragma unroll
    for (int e = 0; e < NEXP; ++e) acc[e] += xv * gw[e * HID + h];
  }
#pragma unroll
  for (int e = 0; e < NEXP; ++e)
    for (int m = 16; m >= 1; m >>= 1) acc[e] += __shfl_xor(acc[e], m, 32);

  if (lane == 0) {
    float mx = acc[0];
#pragma unroll
    for (int e = 1; e < NEXP; ++e) mx = fmaxf(mx, acc[e]);
    float p[NEXP], s = 0.0f;
#pragma unroll
    for (int e = 0; e < NEXP; ++e) { p[e] = __expf(acc[e] - mx); s += p[e]; }
    float inv = 1.0f / s;
#pragma unroll
    for (int e = 0; e < NEXP; ++e) p[e] *= inv;
    int i1 = 0;
#pragma unroll
    for (int e = 1; e < NEXP; ++e) if (p[e] > p[i1]) i1 = e;   // first-on-tie
    int i2 = (i1 == 0) ? 1 : 0;
#pragma unroll
    for (int e = 0; e < NEXP; ++e)
      if (e != i1 && e != i2 && p[e] > p[i2]) i2 = e;
    float d = 1.0f / (p[i1] + p[i2]);
    float w[NEXP];
#pragma unroll
    for (int e = 0; e < NEXP; ++e) w[e] = 0.0f;
    w[i1] = p[i1] * d;
    w[i2] = p[i2] * d;
#pragma unroll
    for (int e = 0; e < NEXP; ++e) combine[(size_t)tok * NEXP + e] = w[e];
  }
}

// ---- fused gate/up GEMM + SwiGLU (+ optional per-token combine scale) -----
// act[m,n] = silu(X Wg^T)[m,n] * (X Wu^T)[m,n] * (combine ? combine[m,e] : 1)
// X:[T,K] fp32, Wg/Wu:[N,K] fp32, act:[T,N] fp32.

__global__ __launch_bounds__(256)
void moe_gate_up_swiglu(const float* __restrict__ X,
                        const float* __restrict__ Wg,
                        const float* __restrict__ Wu,
                        float* __restrict__ act, int K, int N,
                        const float* __restrict__ combine, int eIdx) {
  __shared__ __align__(16) unsigned short As[BM * LDT];
  __shared__ __align__(16) unsigned short Gs[BN * LDT];
  __shared__ __align__(16) unsigned short Us[BN * LDT];

  const int t    = threadIdx.x;
  const int lane = t & 31;
  const int wave = t >> 5;
  const int wm   = wave >> 1;               // 0..3 -> wave M offset *32
  const int wn   = wave & 1;                // 0..1 -> wave N offset *32
  const int row0 = blockIdx.y * BM;
  const int col0 = blockIdx.x * BN;

  // per-thread cooperative-load coordinates
  const int ra = t >> 3, ca = (t & 7) * 4;        // X: 4 chunks of (row+32k)
  const int rw = t >> 3, cw = (t & 7) * 4;        // W: 2 chunks of (row+32k)

  v8f accg[2][2] = {};
  v8f accu[2][2] = {};

  // prefetch tile k0 = 0 into registers
  float4 xa[4], wg[2], wu[2];
#pragma unroll
  for (int i = 0; i < 4; ++i)
    xa[i] = *(const float4*)(X + (size_t)(row0 + ra + i * 32) * K + ca);
#pragma unroll
  for (int i = 0; i < 2; ++i) {
    size_t off = (size_t)(col0 + rw + i * 32) * K + cw;
    wg[i] = *(const float4*)(Wg + off);
    wu[i] = *(const float4*)(Wu + off);
  }

  for (int k0 = 0; k0 < K; k0 += KC) {
    // commit prefetched tile to LDS as bf16
#pragma unroll
    for (int i = 0; i < 4; ++i) store_bf4(&As[(ra + i * 32) * LDT + ca], xa[i]);
#pragma unroll
    for (int i = 0; i < 2; ++i) {
      store_bf4(&Gs[(rw + i * 32) * LDT + cw], wg[i]);
      store_bf4(&Us[(rw + i * 32) * LDT + cw], wu[i]);
    }
    __syncthreads();

    // prefetch next tile while WMMAs run from LDS
    if (k0 + KC < K) {
      int kn = k0 + KC;
#pragma unroll
      for (int i = 0; i < 4; ++i)
        xa[i] = *(const float4*)(X + (size_t)(row0 + ra + i * 32) * K + kn + ca);
#pragma unroll
      for (int i = 0; i < 2; ++i) {
        size_t off = (size_t)(col0 + rw + i * 32) * K + kn + cw;
        wg[i] = *(const float4*)(Wg + off);
        wu[i] = *(const float4*)(Wu + off);
      }
    }

    v16bf a0 = load_frag(&As[(wm * 32 +  0) * LDT], lane);
    v16bf a1 = load_frag(&As[(wm * 32 + 16) * LDT], lane);
    v16bf g0 = load_frag(&Gs[(wn * 32 +  0) * LDT], lane);
    v16bf g1 = load_frag(&Gs[(wn * 32 + 16) * LDT], lane);
    v16bf u0 = load_frag(&Us[(wn * 32 +  0) * LDT], lane);
    v16bf u1 = load_frag(&Us[(wn * 32 + 16) * LDT], lane);

    accg[0][0] = wmma_bf16(a0, g0, accg[0][0]);
    accg[0][1] = wmma_bf16(a0, g1, accg[0][1]);
    accg[1][0] = wmma_bf16(a1, g0, accg[1][0]);
    accg[1][1] = wmma_bf16(a1, g1, accg[1][1]);
    accu[0][0] = wmma_bf16(a0, u0, accu[0][0]);
    accu[0][1] = wmma_bf16(a0, u1, accu[0][1]);
    accu[1][0] = wmma_bf16(a1, u0, accu[1][0]);
    accu[1][1] = wmma_bf16(a1, u1, accu[1][1]);
    __syncthreads();
  }

  // C/D layout: lane holds col (lane&15); rows = (lane>>4)*8 + vgpr.
  const int cn = lane & 15;
  const int rh = (lane >> 4) << 3;
#pragma unroll
  for (int mf = 0; mf < 2; ++mf)
#pragma unroll
    for (int nf = 0; nf < 2; ++nf)
#pragma unroll
      for (int v = 0; v < 8; ++v) {
        int m = row0 + wm * 32 + mf * 16 + rh + v;
        int n = col0 + wn * 32 + nf * 16 + cn;
        float g = accg[mf][nf][v];
        float u = accu[mf][nf][v];
        float s = fast_silu(g) * u;
        if (combine) s *= combine[(size_t)m * NEXP + eIdx];
        act[(size_t)m * N + n] = s;
      }
}

// ---- down projection: out[T,N] (=|+=) act[T,K] @ Wd[N,K]^T ----------------

__global__ __launch_bounds__(256)
void moe_down(const float* __restrict__ A, const float* __restrict__ W,
              float* __restrict__ out, int K, int N, int accum) {
  __shared__ __align__(16) unsigned short As[BM * LDT];
  __shared__ __align__(16) unsigned short Ws[BN * LDT];

  const int t    = threadIdx.x;
  const int lane = t & 31;
  const int wave = t >> 5;
  const int wm   = wave >> 1;
  const int wn   = wave & 1;
  const int row0 = blockIdx.y * BM;
  const int col0 = blockIdx.x * BN;

  const int ra = t >> 3, ca = (t & 7) * 4;
  const int rw = t >> 3, cw = (t & 7) * 4;

  v8f acc[2][2] = {};

  float4 xa[4], wb[2];
#pragma unroll
  for (int i = 0; i < 4; ++i)
    xa[i] = *(const float4*)(A + (size_t)(row0 + ra + i * 32) * K + ca);
#pragma unroll
  for (int i = 0; i < 2; ++i)
    wb[i] = *(const float4*)(W + (size_t)(col0 + rw + i * 32) * K + cw);

  for (int k0 = 0; k0 < K; k0 += KC) {
#pragma unroll
    for (int i = 0; i < 4; ++i) store_bf4(&As[(ra + i * 32) * LDT + ca], xa[i]);
#pragma unroll
    for (int i = 0; i < 2; ++i) store_bf4(&Ws[(rw + i * 32) * LDT + cw], wb[i]);
    __syncthreads();

    if (k0 + KC < K) {
      int kn = k0 + KC;
#pragma unroll
      for (int i = 0; i < 4; ++i)
        xa[i] = *(const float4*)(A + (size_t)(row0 + ra + i * 32) * K + kn + ca);
#pragma unroll
      for (int i = 0; i < 2; ++i)
        wb[i] = *(const float4*)(W + (size_t)(col0 + rw + i * 32) * K + kn + cw);
    }

    v16bf a0 = load_frag(&As[(wm * 32 +  0) * LDT], lane);
    v16bf a1 = load_frag(&As[(wm * 32 + 16) * LDT], lane);
    v16bf b0 = load_frag(&Ws[(wn * 32 +  0) * LDT], lane);
    v16bf b1 = load_frag(&Ws[(wn * 32 + 16) * LDT], lane);

    acc[0][0] = wmma_bf16(a0, b0, acc[0][0]);
    acc[0][1] = wmma_bf16(a0, b1, acc[0][1]);
    acc[1][0] = wmma_bf16(a1, b0, acc[1][0]);
    acc[1][1] = wmma_bf16(a1, b1, acc[1][1]);
    __syncthreads();
  }

  const int cn = lane & 15;
  const int rh = (lane >> 4) << 3;
#pragma unroll
  for (int mf = 0; mf < 2; ++mf)
#pragma unroll
    for (int nf = 0; nf < 2; ++nf)
#pragma unroll
      for (int v = 0; v < 8; ++v) {
        int m = row0 + wm * 32 + mf * 16 + rh + v;
        int n = col0 + wn * 32 + nf * 16 + cn;
        size_t idx = (size_t)m * N + n;
        float r = acc[mf][nf][v];
        out[idx] = accum ? (out[idx] + r) : r;   // same-stream launches serialize
      }
}

// ---- launch ---------------------------------------------------------------

extern "C" void kernel_launch(void* const* d_in, const int* in_sizes, int n_in,
                              void* d_out, int out_size, void* d_ws,
                              size_t ws_size, hipStream_t stream) {
  (void)in_sizes; (void)n_in; (void)out_size; (void)ws_size;

  const float* x  = (const float*)d_in[0];   // [T, H]
  const float* gw = (const float*)d_in[1];   // [E, H]
  const float* eg = (const float*)d_in[2];   // [E, I, H]
  const float* eu = (const float*)d_in[3];   // [E, I, H]
  const float* ed = (const float*)d_in[4];   // [E, H, I]
  const float* sg = (const float*)d_in[5];   // [IS, H]
  const float* su = (const float*)d_in[6];   // [IS, H]
  const float* sd = (const float*)d_in[7];   // [H, IS]
  float* out = (float*)d_out;                // [T, H]

  // workspace: combine [T,E] fp32 (64 KB) then act [T, IS] fp32 (44 MB)
  float* combine = (float*)d_ws;
  float* act = (float*)((char*)d_ws + (size_t)T_TOK * NEXP * sizeof(float));

  // 1) routing: dense combine weights (zero for unselected experts)
  moe_router<<<T_TOK / 8, 256, 0, stream>>>(x, gw, combine);

  // 2) shared expert: swiglu then down (plain write initializes d_out)
  moe_gate_up_swiglu<<<dim3(ISH / BN, T_TOK / BM), 256, 0, stream>>>(
      x, sg, su, act, HID, ISH, nullptr, 0);
  moe_down<<<dim3(HID / BN, T_TOK / BM), 256, 0, stream>>>(
      act, sd, out, ISH, HID, /*accum=*/0);

  // 3) routed experts: combine weight folded into the intermediate (linear),
  //    down-proj accumulates; act scratch reused (same-stream ordering).
  for (int e = 0; e < NEXP; ++e) {
    moe_gate_up_swiglu<<<dim3(IEXP / BN, T_TOK / BM), 256, 0, stream>>>(
        x, eg + (size_t)e * IEXP * HID, eu + (size_t)e * IEXP * HID, act,
        HID, IEXP, combine, e);
    moe_down<<<dim3(HID / BN, T_TOK / BM), 256, 0, stream>>>(
        act, ed + (size_t)e * HID * IEXP, out, IEXP, HID, /*accum=*/1);
  }
}
